// SelfAttnReduction_47132971106614
// MI455X (gfx1250) — matrix operand order repaired and checked by reading, not compiled, not run on hardware
//
#include <hip/hip_runtime.h>
#include <math.h>

// ---------------------------------------------------------------------------
// SelfAttnReduction for MI455X (gfx1250, wave32, WMMA).
//
// Roofline: ~1.0e11 FLOPs in 3 GEMMs ([65536x512] x [512x512]) vs ~192MB of
// HBM traffic -> matrix-core bound.  x/W are converted to bf16 once and the
// GEMMs run on v_wmma_f32_16x16x32_bf16.  Register blocking: each wave owns a
// 16x64 output tile so a single A fragment feeds 4 WMMAs.  The K-loop uses
// two explicit fragment buffers in an unroll-by-2 steady state: each buffer
// is always re-LOADED (never copied), so the loads run one full k-step ahead
// of their consuming WMMAs with no v_mov rotation overhead.  bf16 x copy
// (64MB) + weights live in the 192MB L2, so tile re-reads stay on-chip.
// ---------------------------------------------------------------------------

#define BB 16
#define LL 4096
#define DD 512
#define BL (BB * LL)

typedef __bf16 bf16_t;
typedef __bf16 v16bf __attribute__((ext_vector_type(16)));
typedef __bf16 v8bf  __attribute__((ext_vector_type(8)));
typedef __bf16 v4bf  __attribute__((ext_vector_type(4)));
typedef float  v8f   __attribute__((ext_vector_type(8)));

union Frag16 { v16bf v; v8bf h[2]; };

// A fragment (16x32 MxK, bf16): lane row = m0+(lane&15); hi=(lane>>4);
// elems 0..7 hold K=k0+hi*8..+7, elems 8..15 hold K=k0+16+hi*8..+7 (ISA 7.12.2).
__device__ __forceinline__ v16bf load_a(const bf16_t* __restrict__ row, int k0, int hi) {
  Frag16 f;
  f.h[0] = *(const v8bf*)(row + k0 + hi * 8);
  f.h[1] = *(const v8bf*)(row + k0 + 16 + hi * 8);
  return f.v;
}

// B fragment (32x16 KxN, bf16): lane col = n0+(lane&15); hi=(lane>>4);
// elems 0..7 hold K=k0+hi*8..+7, elems 8..15 hold K=k0+16+hi*8..+7 (same as A).
// B[k][n] = W[n*DD + k] -> contiguous in k for fixed n (W stored [e][d]).
__device__ __forceinline__ v16bf load_b(const bf16_t* __restrict__ col, int k0, int hi) {
  Frag16 f;
  const bf16_t* p = col + k0 + hi * 8;
  f.h[0] = *(const v8bf*)(p);
  f.h[1] = *(const v8bf*)(p + 16);
  return f.v;
}

__device__ __forceinline__ v8f wmma_bf16(v16bf a, v16bf b, v8f c) {
  // D = A*B + C, f32 accumulate. 8 args: neg_a, A, neg_b, B, c_mod, C, reuse_a, reuse_b
  return __builtin_amdgcn_wmma_f32_16x16x32_bf16(false, a, false, b, (short)0, c,
                                                 false, false);
}

// ---------------------------------------------------------------------------
__global__ void zero_f32(float* p, int n) {
  int i = blockIdx.x * blockDim.x + threadIdx.x;
  if (i < n) p[i] = 0.0f;
}

__global__ void cvt_bf16(const float* __restrict__ src, bf16_t* __restrict__ dst, int n4) {
  int i = blockIdx.x * blockDim.x + threadIdx.x;
  int stride = gridDim.x * blockDim.x;
  for (; i < n4; i += stride) {
    float4 f = ((const float4*)src)[i];
    v4bf o = { (__bf16)f.x, (__bf16)f.y, (__bf16)f.z, (__bf16)f.w };
    ((v4bf*)dst)[i] = o;
  }
}

// ---------------------------------------------------------------------------
// Q pass: per wave a 16(M)x64(N) tile (4 WMMA accumulators sharing each A
// fragment), double-buffered K-loop over 512.  After tanh, columns are
// reduced over the 16 rows (shfl_xor 16), across the 8 waves via LDS, then
// one f32 atomicAdd per column into q_sum[b*DD+n].
// grid = (BL/128, DD/64), block = 256 (8 waves, each 16 rows x 64 cols).
__global__ __launch_bounds__(256) void qpass_kernel(
    const bf16_t* __restrict__ xb, const bf16_t* __restrict__ Wqb,
    const float* __restrict__ bq, float* __restrict__ q_sum) {
  const int lane = threadIdx.x & 31;
  const int wv   = threadIdx.x >> 5;
  const int lo   = lane & 15;
  const int hi   = lane >> 4;
  const int m0   = blockIdx.x * 128 + wv * 16;
  const int n0   = blockIdx.y * 64;
  const int b    = m0 >> 12;  // LL == 4096

  const bf16_t* arow  = xb  + (size_t)(m0 + lo) * DD;
  const bf16_t* bcol0 = Wqb + (size_t)(n0 + lo) * DD;  // sub-tile s at +s*16*DD

  v8f c0 = {}, c1 = {}, c2 = {}, c3 = {};

  // Buffer X preloaded with k = 0.
  v16bf aX  = load_a(arow, 0, hi);
  v16bf b0X = load_b(bcol0,           0, hi);
  v16bf b1X = load_b(bcol0 + 16 * DD, 0, hi);
  v16bf b2X = load_b(bcol0 + 32 * DD, 0, hi);
  v16bf b3X = load_b(bcol0 + 48 * DD, 0, hi);
  v16bf aY, b0Y, b1Y, b2Y, b3Y;

#pragma unroll 1
  for (int k0 = 0; k0 < DD - 64; k0 += 64) {
    // Load buffer Y (k0+32) while WMMAs consume buffer X (k0).
    aY  = load_a(arow, k0 + 32, hi);
    b0Y = load_b(bcol0,           k0 + 32, hi);
    b1Y = load_b(bcol0 + 16 * DD, k0 + 32, hi);
    b2Y = load_b(bcol0 + 32 * DD, k0 + 32, hi);
    b3Y = load_b(bcol0 + 48 * DD, k0 + 32, hi);
    c0 = wmma_bf16(aX, b0X, c0);
    c1 = wmma_bf16(aX, b1X, c1);
    c2 = wmma_bf16(aX, b2X, c2);
    c3 = wmma_bf16(aX, b3X, c3);
    // Re-load buffer X (k0+64) while WMMAs consume buffer Y (k0+32).
    aX  = load_a(arow, k0 + 64, hi);
    b0X = load_b(bcol0,           k0 + 64, hi);
    b1X = load_b(bcol0 + 16 * DD, k0 + 64, hi);
    b2X = load_b(bcol0 + 32 * DD, k0 + 64, hi);
    b3X = load_b(bcol0 + 48 * DD, k0 + 64, hi);
    c0 = wmma_bf16(aY, b0Y, c0);
    c1 = wmma_bf16(aY, b1Y, c1);
    c2 = wmma_bf16(aY, b2Y, c2);
    c3 = wmma_bf16(aY, b3Y, c3);
  }
  // Tail: k-steps DD-64 (in buffer X) and DD-32 (loaded into Y now).
  aY  = load_a(arow, DD - 32, hi);
  b0Y = load_b(bcol0,           DD - 32, hi);
  b1Y = load_b(bcol0 + 16 * DD, DD - 32, hi);
  b2Y = load_b(bcol0 + 32 * DD, DD - 32, hi);
  b3Y = load_b(bcol0 + 48 * DD, DD - 32, hi);
  c0 = wmma_bf16(aX, b0X, c0);
  c1 = wmma_bf16(aX, b1X, c1);
  c2 = wmma_bf16(aX, b2X, c2);
  c3 = wmma_bf16(aX, b3X, c3);
  c0 = wmma_bf16(aY, b0Y, c0);
  c1 = wmma_bf16(aY, b1Y, c1);
  c2 = wmma_bf16(aY, b2Y, c2);
  c3 = wmma_bf16(aY, b3Y, c3);

  float colsum[4];
  {
    v8f* cs[4] = {&c0, &c1, &c2, &c3};
#pragma unroll
    for (int s = 0; s < 4; ++s) {
      const float bias = bq[n0 + s * 16 + lo];
      float t = 0.0f;
#pragma unroll
      for (int r = 0; r < 8; ++r) t += tanhf((*cs[s])[r] + bias);
      t += __shfl_xor(t, 16, 32);  // lane^16 holds same column, other 8 rows
      colsum[s] = t;
    }
  }

  __shared__ float red[8][64];
  if (hi == 0) {
#pragma unroll
    for (int s = 0; s < 4; ++s) red[wv][s * 16 + lo] = colsum[s];
  }
  __syncthreads();
  if (wv == 0) {
#pragma unroll
    for (int j = 0; j < 2; ++j) {
      const int col = lane + j * 32;
      float s = 0.0f;
#pragma unroll
      for (int i = 0; i < 8; ++i) s += red[i][col];
      atomicAdd(&q_sum[b * DD + n0 + col], s);
    }
  }
}

// ---------------------------------------------------------------------------
// K/V pass: each wave owns 16 rows and loops over N in chunks of 32 columns;
// per K-step one A fragment feeds 4 WMMAs (2 Wk sub-tiles + 2 Wv sub-tiles),
// double-buffered as above.
// Produces k_dot[row] = tanh(x Wk^T)[row,:] . q_sum[b,:]  and
//          v_sum[row] = sum_n tanh(x Wv^T)[row,n].
// grid = BL/128, block = 256.
__global__ __launch_bounds__(256) void kvpass_kernel(
    const bf16_t* __restrict__ xb, const bf16_t* __restrict__ Wkb,
    const bf16_t* __restrict__ Wvb, const float* __restrict__ bkv,
    const float* __restrict__ bvv, const float* __restrict__ q_sum,
    float* __restrict__ k_dot, float* __restrict__ v_sum) {
  const int lane = threadIdx.x & 31;
  const int wv   = threadIdx.x >> 5;
  const int lo   = lane & 15;
  const int hi   = lane >> 4;
  const int m0   = blockIdx.x * 128 + wv * 16;
  const int b    = m0 >> 12;

  const bf16_t* arow = xb + (size_t)(m0 + lo) * DD;

  float accK[8] = {0, 0, 0, 0, 0, 0, 0, 0};
  float accV[8] = {0, 0, 0, 0, 0, 0, 0, 0};

#pragma unroll 1
  for (int n0 = 0; n0 < DD; n0 += 32) {
    const bf16_t* bk0 = Wkb + (size_t)(n0 + lo) * DD;  // +16*DD = next sub-tile
    const bf16_t* bv0 = Wvb + (size_t)(n0 + lo) * DD;
    __builtin_prefetch(bk0 + 32 * DD, 0, 1);  // next N-chunk's weight column

    v8f ck0 = {}, ck1 = {}, cv0 = {}, cv1 = {};

    // Buffer X preloaded with k = 0.
    v16bf aX   = load_a(arow, 0, hi);
    v16bf fk0X = load_b(bk0,           0, hi);
    v16bf fk1X = load_b(bk0 + 16 * DD, 0, hi);
    v16bf fv0X = load_b(bv0,           0, hi);
    v16bf fv1X = load_b(bv0 + 16 * DD, 0, hi);
    v16bf aY, fk0Y, fk1Y, fv0Y, fv1Y;

#pragma unroll 1
    for (int k0 = 0; k0 < DD - 64; k0 += 64) {
      aY   = load_a(arow, k0 + 32, hi);
      fk0Y = load_b(bk0,           k0 + 32, hi);
      fk1Y = load_b(bk0 + 16 * DD, k0 + 32, hi);
      fv0Y = load_b(bv0,           k0 + 32, hi);
      fv1Y = load_b(bv0 + 16 * DD, k0 + 32, hi);
      ck0 = wmma_bf16(aX, fk0X, ck0);
      ck1 = wmma_bf16(aX, fk1X, ck1);
      cv0 = wmma_bf16(aX, fv0X, cv0);
      cv1 = wmma_bf16(aX, fv1X, cv1);
      aX   = load_a(arow, k0 + 64, hi);
      fk0X = load_b(bk0,           k0 + 64, hi);
      fk1X = load_b(bk0 + 16 * DD, k0 + 64, hi);
      fv0X = load_b(bv0,           k0 + 64, hi);
      fv1X = load_b(bv0 + 16 * DD, k0 + 64, hi);
      ck0 = wmma_bf16(aY, fk0Y, ck0);
      ck1 = wmma_bf16(aY, fk1Y, ck1);
      cv0 = wmma_bf16(aY, fv0Y, cv0);
      cv1 = wmma_bf16(aY, fv1Y, cv1);
    }
    // Tail: k-steps DD-64 (buffer X) and DD-32 (loaded into Y now).
    aY   = load_a(arow, DD - 32, hi);
    fk0Y = load_b(bk0,           DD - 32, hi);
    fk1Y = load_b(bk0 + 16 * DD, DD - 32, hi);
    fv0Y = load_b(bv0,           DD - 32, hi);
    fv1Y = load_b(bv0 + 16 * DD, DD - 32, hi);
    ck0 = wmma_bf16(aX, fk0X, ck0);
    ck1 = wmma_bf16(aX, fk1X, ck1);
    cv0 = wmma_bf16(aX, fv0X, cv0);
    cv1 = wmma_bf16(aX, fv1X, cv1);
    ck0 = wmma_bf16(aY, fk0Y, ck0);
    ck1 = wmma_bf16(aY, fk1Y, ck1);
    cv0 = wmma_bf16(aY, fv0Y, cv0);
    cv1 = wmma_bf16(aY, fv1Y, cv1);

    const float q0 = q_sum[b * DD + n0 + lo];
    const float q1 = q_sum[b * DD + n0 + 16 + lo];
    const float bkb0 = bkv[n0 + lo],      bkb1 = bkv[n0 + 16 + lo];
    const float bvb0 = bvv[n0 + lo],      bvb1 = bvv[n0 + 16 + lo];
#pragma unroll
    for (int r = 0; r < 8; ++r) {
      accK[r] += tanhf(ck0[r] + bkb0) * q0 + tanhf(ck1[r] + bkb1) * q1;
      accV[r] += tanhf(cv0[r] + bvb0)      + tanhf(cv1[r] + bvb1);
    }
  }

  // Row reduce over the 16 columns (lane&15 bits); hi bit untouched.
#pragma unroll
  for (int m = 1; m <= 8; m <<= 1) {
#pragma unroll
    for (int r = 0; r < 8; ++r) {
      accK[r] += __shfl_xor(accK[r], m, 32);
      accV[r] += __shfl_xor(accV[r], m, 32);
    }
  }
  if (lo == 0) {  // lanes 0 (rows m0..m0+7) and 16 (rows m0+8..m0+15)
#pragma unroll
    for (int r = 0; r < 8; ++r) {
      const int row = m0 + r + hi * 8;
      k_dot[row] = accK[r];
      v_sum[row] = accV[r];
    }
  }
}

// ---------------------------------------------------------------------------
// Masked softmax over L per batch, fused with the v_sum scaling.
__global__ __launch_bounds__(256) void softmax_kernel(
    const float* __restrict__ k_dot, const float* __restrict__ v_sum,
    const int* __restrict__ lens, float* __restrict__ out) {
  const int b   = blockIdx.x;
  const int len = lens[b];
  const float scale = 0.044194173824159216f;  // 1/sqrt(512)
  __shared__ float sred[256];

  float mx = -INFINITY;
  for (int l = threadIdx.x; l < len; l += 256)
    mx = fmaxf(mx, k_dot[b * LL + l] * scale);
  sred[threadIdx.x] = mx;
  __syncthreads();
  for (int s = 128; s > 0; s >>= 1) {
    if (threadIdx.x < s)
      sred[threadIdx.x] = fmaxf(sred[threadIdx.x], sred[threadIdx.x + s]);
    __syncthreads();
  }
  mx = sred[0];
  __syncthreads();

  float sum = 0.0f;
  for (int l = threadIdx.x; l < len; l += 256)
    sum += expf(k_dot[b * LL + l] * scale - mx);
  sred[threadIdx.x] = sum;
  __syncthreads();
  for (int s = 128; s > 0; s >>= 1) {
    if (threadIdx.x < s) sred[threadIdx.x] += sred[threadIdx.x + s];
    __syncthreads();
  }
  const float inv = 1.0f / sred[0];

  for (int l = threadIdx.x; l < LL; l += 256) {
    float o = 0.0f;
    if (l < len)
      o = v_sum[b * LL + l] * expf(k_dot[b * LL + l] * scale - mx) * inv;
    out[b * LL + l] = o;
  }
}

// ---------------------------------------------------------------------------
extern "C" void kernel_launch(void* const* d_in, const int* in_sizes, int n_in,
                              void* d_out, int out_size, void* d_ws, size_t ws_size,
                              hipStream_t stream) {
  (void)in_sizes; (void)n_in; (void)out_size; (void)ws_size;
  const float* x    = (const float*)d_in[0];
  const float* Wk   = (const float*)d_in[1];
  const float* bk   = (const float*)d_in[2];
  const float* Wq   = (const float*)d_in[3];
  const float* bq   = (const float*)d_in[4];
  const float* Wv   = (const float*)d_in[5];
  const float* bv   = (const float*)d_in[6];
  const int*   lens = (const int*)d_in[7];
  float* out = (float*)d_out;

  // Workspace layout (all offsets 4KB-aligned). Total ~69.2 MB.
  char* ws = (char*)d_ws;
  const size_t OFF_XB  = 0;                                   // BL*DD bf16 = 64MB
  const size_t OFF_WKB = OFF_XB  + (size_t)BL * DD * 2;       // DD*DD bf16
  const size_t OFF_WQB = OFF_WKB + (size_t)DD * DD * 2;
  const size_t OFF_WVB = OFF_WQB + (size_t)DD * DD * 2;
  const size_t OFF_QS  = OFF_WVB + (size_t)DD * DD * 2;       // BB*DD f32
  const size_t OFF_KD  = OFF_QS  + (size_t)BB * DD * 4;       // BL f32
  const size_t OFF_VS  = OFF_KD  + (size_t)BL * 4;            // BL f32
  bf16_t* xb    = (bf16_t*)(ws + OFF_XB);
  bf16_t* Wkb   = (bf16_t*)(ws + OFF_WKB);
  bf16_t* Wqb   = (bf16_t*)(ws + OFF_WQB);
  bf16_t* Wvb   = (bf16_t*)(ws + OFF_WVB);
  float*  q_sum = (float*)(ws + OFF_QS);
  float*  k_dot = (float*)(ws + OFF_KD);
  float*  v_sum = (float*)(ws + OFF_VS);

  // 1) q_sum <- 0 (atomics accumulate into it each call; ws is not re-zeroed).
  zero_f32<<<(BB * DD + 255) / 256, 256, 0, stream>>>(q_sum, BB * DD);

  // 2) bf16 conversions (x once; weights are tiny).
  cvt_bf16<<<4096, 256, 0, stream>>>(x,  xb,  (BL * DD) / 4);
  cvt_bf16<<<256,  256, 0, stream>>>(Wk, Wkb, (DD * DD) / 4);
  cvt_bf16<<<256,  256, 0, stream>>>(Wq, Wqb, (DD * DD) / 4);
  cvt_bf16<<<256,  256, 0, stream>>>(Wv, Wvb, (DD * DD) / 4);

  // 3) Q pass: q_sum[b,d] = sum_l tanh(x Wq^T)[b,l,d]
  qpass_kernel<<<dim3(BL / 128, DD / 64), 256, 0, stream>>>(xb, Wqb, bq, q_sum);

  // 4) K/V pass: k_dot and v_sum per row.
  kvpass_kernel<<<BL / 128, 256, 0, stream>>>(xb, Wkb, Wvb, bk, bv, q_sum,
                                              k_dot, v_sum);

  // 5) Masked softmax + scale.
  softmax_kernel<<<BB, 256, 0, stream>>>(k_dot, v_sum, lens, out);
}
